// NERTagger_74113955660355
// MI455X (gfx1250) — matrix operand order repaired and checked by reading, not compiled, not run on hardware
//
#include <hip/hip_runtime.h>
#include <hip/hip_bf16.h>
#include <stdint.h>

typedef __attribute__((ext_vector_type(16))) _Float16 v16h;
typedef __attribute__((ext_vector_type(8)))  _Float16 v8h;
typedef __attribute__((ext_vector_type(8)))  float    v8f;

#define BS 1024
#define SL 512
#define NC 50
#define NCP 64  // NC padded to 64 for WMMA K and N

// ---------------------------------------------------------------------------
// Kernel 1: per-batch unary+binary score. One block per batch row, 256-thread
// reduction over the 512 timesteps (fully parallel, deterministic).
// ---------------------------------------------------------------------------
__global__ void crf_side_kernel(const float* __restrict__ inp,
                                const float* __restrict__ trans,
                                const int*   __restrict__ masks,
                                const int*   __restrict__ tags,
                                float* __restrict__ ub) {
    __shared__ float buf[256];
    const int b   = blockIdx.x;
    const int tid = threadIdx.x;
    float u = 0.0f;
    for (int t = tid; t < SL; t += 256) {
        int m  = masks[(size_t)b * SL + t];
        int tg = tags[(size_t)b * SL + t];
        if (!m) {
            u += inp[((size_t)b * SL + t) * NC + tg];            // unary
            if (t > 0) {
                int pv = tags[(size_t)b * SL + t - 1];
                u += trans[pv * NC + tg];                        // binary
            }
        }
    }
    buf[tid] = u;
    __syncthreads();
    for (int off = 128; off > 0; off >>= 1) {
        if (tid < off) buf[tid] += buf[tid + off];
        __syncthreads();
    }
    if (tid == 0) ub[b] = buf[0];
}

// ---------------------------------------------------------------------------
// Kernel 1b: copy transitions (second tuple output) into d_out[1 .. 2500]
// ---------------------------------------------------------------------------
__global__ void crf_copy_kernel(const float* __restrict__ trans,
                                float* __restrict__ out) {
    int i = blockIdx.x * blockDim.x + threadIdx.x;
    if (i < NC * NC) out[1 + i] = trans[i];
}

// ---------------------------------------------------------------------------
// Kernel 2: CRF forward scan. One BLOCK per 16-batch tile (grid = 64); the 4
// waves split the N (next-tag) dimension. Two barriers per timestep:
//   [A] row-max (8-lane butterfly) + exp -> f16 staging
//   [B] A-frag loads -> 2x v_wmma_f32_16x16x32_f16 -> straightline log ->
//       branchless bit-select against register-resident alpha column -> store
// ---------------------------------------------------------------------------
__global__ void __launch_bounds__(128)
crf_scan_kernel(const float* __restrict__ inp,
                const float* __restrict__ trans,
                const int*   __restrict__ masks,
                const float* __restrict__ ub,
                float* __restrict__ ll) {
    __shared__ __align__(16) _Float16 sExpT[NCP][NCP];   // exp(T) f16, padded
    __shared__ __align__(16) float    sAlpha[16][NCP];   // alphas for 16 batches
    __shared__ __align__(16) _Float16 sP[16][NCP];       // p = exp(alpha - m)
    __shared__ __align__(16) float    sM[16];            // per-row max
    __shared__ __align__(16) uint32_t sMk[16];           // ~0u if masked, else 0

    const int tid   = threadIdx.x;
    const int lane  = tid & 31;
    const int wave  = tid >> 5;        // = N tile index nt
    const int l16   = lane & 15;
    const int khalf = lane >> 4;
    const int b0    = blockIdx.x * 16;
    const int rrow  = tid >> 3;        // 0..15: row for reduction phase
    const int rseg  = tid & 7;         // 8 threads per row

    // Stage exp(T) into LDS as f16, zero-padded to 64x64.
    for (int i = tid; i < NCP * NCP; i += 128) {
        int r = i >> 6, c = i & 63;
        float v = (r < NC && c < NC) ? __expf(trans[r * NC + c]) : 0.0f;
        sExpT[r][c] = (_Float16)v;
    }
    __syncthreads();

    // Constant B fragments for this wave's N tile (32x16 f16 B layout):
    // element e of fragment kc: K = kc*32 + khalf*16 + e, N = wave*16 + l16.
    v16h Bf0, Bf1;
#pragma unroll
    for (int e = 0; e < 16; ++e) {
        Bf0[e] = sExpT[khalf * 16 + e][wave * 16 + l16];
        Bf1[e] = sExpT[32 + khalf * 16 + e][wave * 16 + l16];
    }

    const int colw  = wave * 16 + l16;            // this lane's output column
    const int colwc = (colw < NC) ? colw : 0;     // clamped for safe loads

    // Register-resident alpha column: av[r] = alpha[r+8*khalf][colw].
    // Each lane owns exactly the elements it produces each step (matches the
    // 16x16 f32 D-matrix VGPR layout). 128 threads x 8 = the full 16x64 tile.
    float av[8];
#pragma unroll
    for (int r = 0; r < 8; ++r) {
        int row = r + khalf * 8;
        float v = inp[((size_t)(b0 + row) * SL) * NC + colwc];
        av[r] = (colw < NC) ? v : -1e30f;
        sAlpha[row][colw] = av[r];
    }
    __syncthreads();

    for (int t = 1; t < SL; ++t) {
        // ---- prefetch this step's emission scores (hides global latency) ----
        float xr[8];
#pragma unroll
        for (int r = 0; r < 8; ++r) {
            int row = r + khalf * 8;
            xr[r] = inp[((size_t)(b0 + row) * SL + t) * NC + colwc];
        }

        // ---- phase A: row max (butterfly: all 8 lanes get it) + exp ----
        float mx = -1e30f;
#pragma unroll
        for (int k = 0; k < 8; ++k)
            mx = fmaxf(mx, sAlpha[rrow][rseg * 8 + k]);
        mx = fmaxf(mx, __shfl_xor(mx, 1, 32));
        mx = fmaxf(mx, __shfl_xor(mx, 2, 32));
        mx = fmaxf(mx, __shfl_xor(mx, 4, 32));
        if (rseg == 0) {
            sM[rrow]  = mx;                                   // for phase B
            sMk[rrow] = masks[(size_t)(b0 + rrow) * SL + t] ? ~0u : 0u;
        }
#pragma unroll
        for (int k = 0; k < 8; ++k)
            sP[rrow][rseg * 8 + k] =
                (_Float16)__expf(sAlpha[rrow][rseg * 8 + k] - mx);
        __syncthreads();

        // ---- phase B: row constants, A frags, WMMA, straightline log ----
        float    mrow[8];
        uint32_t mkrow[8];
#pragma unroll
        for (int r = 0; r < 8; ++r) {
            mrow[r]  = sM[r + khalf * 8];
            mkrow[r] = sMk[r + khalf * 8];
        }

        // A fragments (contiguous K order, matching B): element e:
        // K=kc*32+khalf*16+e.
        const _Float16* prow = &sP[l16][0];
        v8h a = *(const v8h*)(prow + khalf * 16);
        v8h b = *(const v8h*)(prow + 8 + khalf * 16);
        v8h c = *(const v8h*)(prow + 32 + khalf * 16);
        v8h d = *(const v8h*)(prow + 40 + khalf * 16);
        v16h A0, A1;
#pragma unroll
        for (int e = 0; e < 8; ++e) {
            A0[e] = a[e];  A0[e + 8] = b[e];
            A1[e] = c[e];  A1[e + 8] = d[e];
        }

        v8f acc = {};
        acc = __builtin_amdgcn_wmma_f32_16x16x32_f16(
            false, A0, false, Bf0, (short)0, acc, false, false);
        acc = __builtin_amdgcn_wmma_f32_16x16x32_f16(
            false, A1, false, Bf1, (short)0, acc, false, false);

        // Straightline: padded columns get acc==0 -> log(0) = -inf -> nv=-inf
        // (m stays finite, exp(-inf-m)=0 next step). Mask handled by a
        // branchless bit-select (keep is ~0u or 0u) -> compiler cannot
        // re-branch and sink the log chains.
#pragma unroll
        for (int r = 0; r < 8; ++r) {
            float nv = xr[r] + __logf(acc[r]) + mrow[r];
            asm volatile("" : "+v"(nv));          // pin: force unconditional
            uint32_t keep = mkrow[r];
            uint32_t sel  = (__float_as_uint(av[r]) & keep) |
                            (__float_as_uint(nv) & ~keep);
            av[r] = __uint_as_float(sel);
            sAlpha[r + khalf * 8][colw] = av[r];
        }
        __syncthreads();
    }

    // ---- final: log_norm[b] = logsumexp_j alpha[b][j]; ll = ub - log_norm ----
    float mx = -1e30f;
#pragma unroll
    for (int k = 0; k < 8; ++k)
        mx = fmaxf(mx, sAlpha[rrow][rseg * 8 + k]);
    mx = fmaxf(mx, __shfl_xor(mx, 1, 32));
    mx = fmaxf(mx, __shfl_xor(mx, 2, 32));
    mx = fmaxf(mx, __shfl_xor(mx, 4, 32));
    float s = 0.0f;
#pragma unroll
    for (int k = 0; k < 8; ++k)
        s += __expf(sAlpha[rrow][rseg * 8 + k] - mx);
    s += __shfl_xor(s, 1, 32);
    s += __shfl_xor(s, 2, 32);
    s += __shfl_xor(s, 4, 32);
    if (rseg == 0) {
        float ln = mx + __logf(s);
        ll[b0 + rrow] = ub[b0 + rrow] - ln;
    }
}

// ---------------------------------------------------------------------------
// Kernel 3: deterministic single-block reduction -> loss = -sum(ll)/BS
// ---------------------------------------------------------------------------
__global__ void crf_reduce_kernel(const float* __restrict__ ll,
                                  float* __restrict__ out) {
    __shared__ float buf[256];
    int tid = threadIdx.x;
    float s = 0.0f;
    for (int i = tid; i < BS; i += 256) s += ll[i];
    buf[tid] = s;
    __syncthreads();
    for (int off = 128; off > 0; off >>= 1) {
        if (tid < off) buf[tid] += buf[tid + off];
        __syncthreads();
    }
    if (tid == 0) out[0] = -buf[0] / (float)BS;
}

extern "C" void kernel_launch(void* const* d_in, const int* in_sizes, int n_in,
                              void* d_out, int out_size, void* d_ws, size_t ws_size,
                              hipStream_t stream) {
    const float* inp   = (const float*)d_in[0];   // (1024, 512, 50) f32
    const float* trans = (const float*)d_in[1];   // (50, 50) f32
    const int*   masks = (const int*)d_in[2];     // (1024, 512) bool->int
    const int*   tags  = (const int*)d_in[3];     // (1024, 512) int32
    float* out = (float*)d_out;                   // [loss, transitions(2500)]
    float* ub  = (float*)d_ws;                    // per-batch unary+binary
    float* ll  = ub + BS;                         // per-batch log-likelihood

    crf_side_kernel<<<BS, 256, 0, stream>>>(inp, trans, masks, tags, ub);
    crf_copy_kernel<<<(NC * NC + 255) / 256, 256, 0, stream>>>(trans, out);
    crf_scan_kernel<<<BS / 16, 128, 0, stream>>>(inp, trans, masks, ub, ll);
    crf_reduce_kernel<<<1, 256, 0, stream>>>(ll, out);
}
